// ConvOffset2d_88132728914543
// MI455X (gfx1250) — matrix operand order repaired and checked by reading, not compiled, not run on hardware
//
#include <hip/hip_runtime.h>

typedef __attribute__((ext_vector_type(16))) __bf16 v16bf;
typedef __attribute__((ext_vector_type(8)))  float  v8f;

// Problem constants (fixed by the reference)
#define NB    8
#define CIN   128
#define HH    64
#define WW    64
#define COUT  256
#define KTAPS 9
#define NPIX  (NB * HH * WW)          // 32768 output pixels
#define KDIM  (CIN * KTAPS)           // 1152
#define NCHUNK (KDIM / 32)            // 36 chunks of K=32

__device__ __forceinline__ unsigned short f2bf(float f) {
  unsigned int u = __builtin_bit_cast(unsigned int, f);
  u += 0x7FFFu + ((u >> 16) & 1u);    // round-to-nearest-even
  return (unsigned short)(u >> 16);
}
__device__ __forceinline__ float bf2f(unsigned short h) {
  return __builtin_bit_cast(float, ((unsigned int)h) << 16);
}

// ---------------------------------------------------------------------------
// Prepass 1: x [B,C,H,W] f32  ->  xh [B,H,W,C] bf16  (LDS-tiled transpose)
// grid: B*H*(C/32) = 8*64*4 = 2048 blocks of 256
// ---------------------------------------------------------------------------
__global__ __launch_bounds__(256)
void dcn_nchw_to_nhwc_bf16(const float* __restrict__ x,
                           unsigned short* __restrict__ xh) {
  __shared__ float tile[32 * 65];     // padded: conflict-free both phases
  const int t   = threadIdx.x;
  const int bid = blockIdx.x;
  const int ct  = bid & 3;            // channel tile (32 ch)
  const int y   = (bid >> 2) & 63;
  const int b   = bid >> 8;
  const int c0  = ct * 32;

  const float* src = x + ((size_t)(b * CIN + c0) * HH + y) * WW;
  #pragma unroll
  for (int j = 0; j < 8; ++j) {
    int e  = t + 256 * j;             // e = i*64 + xc, lanes span xc (coalesced)
    int i  = e >> 6;
    int xc = e & 63;
    tile[i * 65 + xc] = src[(size_t)i * (HH * WW) + xc];
  }
  __syncthreads();

  const int xc = t >> 2;
  const int i0 = (t & 3) * 8;
  unsigned int pk[4];
  #pragma unroll
  for (int j = 0; j < 4; ++j) {
    unsigned short lo = f2bf(tile[(i0 + 2 * j) * 65 + xc]);
    unsigned short hi = f2bf(tile[(i0 + 2 * j + 1) * 65 + xc]);
    pk[j] = (unsigned int)lo | ((unsigned int)hi << 16);
  }
  size_t dst = ((size_t)(b * HH + y) * WW + xc) * CIN + c0 + i0;
  *(uint4*)(xh + dst) = make_uint4(pk[0], pk[1], pk[2], pk[3]);
}

// ---------------------------------------------------------------------------
// Prepass 2: weight [Co,Ci,3,3] f32 -> wh [chunk(36)][co(256)][k(32)] bf16
// chunk = tap*4 + cc ; local k <-> cin = cc*32 + k
// grid: 36*256*32/256 = 1152 blocks of 256
// ---------------------------------------------------------------------------
__global__ __launch_bounds__(256)
void dcn_prep_weight(const float* __restrict__ w,
                     unsigned short* __restrict__ wh) {
  int id    = blockIdx.x * 256 + threadIdx.x;   // < 294912
  int kk    = id & 31;
  int n     = (id >> 5) & 255;
  int chunk = id >> 13;
  int tap   = chunk >> 2;
  int cc    = chunk & 3;
  int cin   = cc * 32 + kk;
  int ky    = tap / 3;
  int kx    = tap - ky * 3;
  wh[id] = f2bf(w[(((size_t)n * CIN + cin) * 3 + ky) * 3 + kx]);
}

// ---------------------------------------------------------------------------
// Main fused kernel: deformable im2col (bf16, in LDS) + WMMA GEMM.
// One block = one output row (b, ho): M-tile = 64 pixels (wo 0..63), N = 256.
// 8 waves: mw = wave&3 (16-pixel subtile), nw = wave>>2 (128-channel half).
// grid: 32768/64 = 512 blocks of 256
// ---------------------------------------------------------------------------
__global__ __launch_bounds__(256)
void dcn_main(const float* __restrict__ off,
              const unsigned short* __restrict__ xh,
              const unsigned short* __restrict__ wh,
              float* __restrict__ out) {
  __shared__ int   sampI[64 * 4];                       // clipped corner bases
  __shared__ float sampW[64 * 4];                       // masked bilinear wts
  __shared__ __align__(16) unsigned short ldsA[64 * 32];   // [pixel][k]  4 KB
  __shared__ __align__(16) unsigned short ldsB[256 * 32];  // [co][k]    16 KB

  const int t    = threadIdx.x;
  const int lane = t & 31;
  const int wv   = t >> 5;
  const int mw   = wv & 3;
  const int nw   = wv >> 2;
  const int sel  = lane >> 4;     // 0: lanes 0-15, 1: lanes 16-31
  const int lr   = lane & 15;

  const int mbase = blockIdx.x * 64;      // one full (b, ho) row
  const int b     = mbase >> 12;
  const int ho    = (mbase >> 6) & 63;

  v8f acc[8];
  #pragma unroll
  for (int i = 0; i < 8; ++i)
    acc[i] = (v8f){0.f, 0.f, 0.f, 0.f, 0.f, 0.f, 0.f, 0.f};

  const float* offB = off + (size_t)b * (2 * KTAPS * HH * WW) + ho * WW;

  for (int tap = 0; tap < KTAPS; ++tap) {
    // ---- per-tap sampling setup (threads 0..63, one per pixel/wo) ----
    if (t < 64) {
      const int wo = t;
      const int ky = tap / 3, kx = tap - 3 * (tap / 3);
      float dy = offB[(2 * tap + 0) * (HH * WW) + wo];
      float dx = offB[(2 * tap + 1) * (HH * WW) + wo];
      float py = (float)(ho - 1 + ky) + dy;
      float px = (float)(wo - 1 + kx) + dx;
      float y0f = floorf(py), x0f = floorf(px);
      int   y0 = (int)y0f,    x0 = (int)x0f;
      float fy = py - y0f,    fx = px - x0f;
      int   y1 = y0 + 1,      x1 = x0 + 1;
      bool vy0 = (y0 >= 0) && (y0 < HH);
      bool vy1 = (y1 >= 0) && (y1 < HH);
      bool vx0 = (x0 >= 0) && (x0 < WW);
      bool vx1 = (x1 >= 0) && (x1 < WW);
      int y0c = y0 < 0 ? 0 : (y0 > HH - 1 ? HH - 1 : y0);
      int y1c = y1 < 0 ? 0 : (y1 > HH - 1 ? HH - 1 : y1);
      int x0c = x0 < 0 ? 0 : (x0 > WW - 1 ? WW - 1 : x0);
      int x1c = x1 < 0 ? 0 : (x1 > WW - 1 ? WW - 1 : x1);
      const int ib = b * (HH * WW * CIN);
      sampI[t * 4 + 0] = ib + (y0c * WW + x0c) * CIN;
      sampI[t * 4 + 1] = ib + (y0c * WW + x1c) * CIN;
      sampI[t * 4 + 2] = ib + (y1c * WW + x0c) * CIN;
      sampI[t * 4 + 3] = ib + (y1c * WW + x1c) * CIN;
      sampW[t * 4 + 0] = (vy0 && vx0) ? (1.f - fy) * (1.f - fx) : 0.f;
      sampW[t * 4 + 1] = (vy0 && vx1) ? (1.f - fy) * fx         : 0.f;
      sampW[t * 4 + 2] = (vy1 && vx0) ? fy * (1.f - fx)         : 0.f;
      sampW[t * 4 + 3] = (vy1 && vx1) ? fy * fx                 : 0.f;
    }
    __syncthreads();

    for (int cc = 0; cc < 4; ++cc) {
      const int chunk = tap * 4 + cc;

      // ---- stage B tile: 256x32 bf16 = 16 KB, fully coalesced ----
      {
        const uint4* src = (const uint4*)(wh + (size_t)chunk * (COUT * 32));
        uint4* dst = (uint4*)ldsB;
        #pragma unroll
        for (int i = 0; i < 4; ++i) dst[t + 256 * i] = src[t + 256 * i];
        if (chunk + 1 < NCHUNK)   // -> global_prefetch_b8 for next chunk
          __builtin_prefetch((const void*)(wh + (size_t)(chunk + 1) * (COUT * 32)), 0, 1);
      }

      // ---- compute A tile: 64 pixels x 32 ch, bilinear in f32 -> bf16 ----
      {
        const int p  = t >> 2;
        const int cs = (t & 3) * 8;
        const int cb = cc * 32 + cs;
        const int i0 = sampI[p * 4 + 0], i1 = sampI[p * 4 + 1];
        const int i2 = sampI[p * 4 + 2], i3 = sampI[p * 4 + 3];
        const float w0 = sampW[p * 4 + 0], w1 = sampW[p * 4 + 1];
        const float w2 = sampW[p * 4 + 2], w3 = sampW[p * 4 + 3];
        uint4 v0 = *(const uint4*)(xh + i0 + cb);
        uint4 v1 = *(const uint4*)(xh + i1 + cb);
        uint4 v2 = *(const uint4*)(xh + i2 + cb);
        uint4 v3 = *(const uint4*)(xh + i3 + cb);
        const unsigned int q0[4] = {v0.x, v0.y, v0.z, v0.w};
        const unsigned int q1[4] = {v1.x, v1.y, v1.z, v1.w};
        const unsigned int q2[4] = {v2.x, v2.y, v2.z, v2.w};
        const unsigned int q3[4] = {v3.x, v3.y, v3.z, v3.w};
        unsigned int pk[4];
        #pragma unroll
        for (int j = 0; j < 4; ++j) {
          float lo = w0 * bf2f((unsigned short)(q0[j] & 0xFFFF))
                   + w1 * bf2f((unsigned short)(q1[j] & 0xFFFF))
                   + w2 * bf2f((unsigned short)(q2[j] & 0xFFFF))
                   + w3 * bf2f((unsigned short)(q3[j] & 0xFFFF));
          float hi = w0 * bf2f((unsigned short)(q0[j] >> 16))
                   + w1 * bf2f((unsigned short)(q1[j] >> 16))
                   + w2 * bf2f((unsigned short)(q2[j] >> 16))
                   + w3 * bf2f((unsigned short)(q3[j] >> 16));
          pk[j] = (unsigned int)f2bf(lo) | ((unsigned int)f2bf(hi) << 16);
        }
        *(uint4*)(ldsA + p * 32 + cs) = make_uint4(pk[0], pk[1], pk[2], pk[3]);
      }
      __syncthreads();

      // ---- fragment loads (contiguous k = 16*sel + s, same in A and B) + 8 WMMAs per wave ----
      union { uint4 q[2]; v16bf v; } Af;
      {
        const int p = mw * 16 + lr;                 // A: lane 0-15 M=lane
        const uint4* ap = (const uint4*)(ldsA + p * 32);
        Af.q[0] = ap[sel * 2 + 0];   // K0..15 (lanes<16) / K16..31 (lanes>=16)
        Af.q[1] = ap[sel * 2 + 1];
      }
      #pragma unroll
      for (int i = 0; i < 8; ++i) {
        const int n = nw * 128 + i * 16 + lr;       // B: N = lane%16
        const uint4* bp = (const uint4*)(ldsB + n * 32);
        union { uint4 q[2]; v16bf v; } Bf;
        Bf.q[0] = bp[sel * 2 + 0];   // K0..15 (lanes<16) / K16..31 (lanes>=16)
        Bf.q[1] = bp[sel * 2 + 1];
        acc[i] = __builtin_amdgcn_wmma_f32_16x16x32_bf16(
            false, Af.v, false, Bf.v, (short)0, acc[i], false, false);
      }
      __syncthreads();
    }
  }

  // ---- epilogue: acc element (m,n): m = r + 8*sel, co = nw*128+i*16+lr ----
  float* outB = out + (size_t)b * (COUT * HH * WW) + (size_t)ho * WW;
  #pragma unroll
  for (int i = 0; i < 8; ++i) {
    const int co = nw * 128 + i * 16 + lr;
    float* op = outB + (size_t)co * (HH * WW) + mw * 16 + sel * 8;
    *(float4*)(op)     = make_float4(acc[i][0], acc[i][1], acc[i][2], acc[i][3]);
    *(float4*)(op + 4) = make_float4(acc[i][4], acc[i][5], acc[i][6], acc[i][7]);
  }
}

// ---------------------------------------------------------------------------
extern "C" void kernel_launch(void* const* d_in, const int* in_sizes, int n_in,
                              void* d_out, int out_size, void* d_ws, size_t ws_size,
                              hipStream_t stream) {
  (void)in_sizes; (void)n_in; (void)out_size; (void)ws_size;
  const float* x   = (const float*)d_in[0];   // [8,128,64,64]
  const float* off = (const float*)d_in[1];   // [8,18,64,64]
  const float* w   = (const float*)d_in[2];   // [256,128,3,3]
  float* out = (float*)d_out;                 // [8,256,64,64]

  unsigned short* xh = (unsigned short*)d_ws;                  // 8.0 MiB
  unsigned short* wh = xh + (size_t)NB * HH * WW * CIN;        // +576 KiB

  dcn_nchw_to_nhwc_bf16<<<NB * HH * (CIN / 32), 256, 0, stream>>>(x, xh);
  dcn_prep_weight<<<(NCHUNK * COUT * 32) / 256, 256, 0, stream>>>(w, wh);
  dcn_main<<<NPIX / 64, 256, 0, stream>>>(off, xh, wh, out);
}
